// Attention_79285096284336
// MI455X (gfx1250) — matrix operand order repaired and checked by reading, not compiled, not run on hardware
//
#include <hip/hip_runtime.h>

// CDNA5 / gfx1250, wave32. bf16 WMMA: D(16x16,f32) = A(16x32,bf16) x B(32x16,bf16) + C
typedef __attribute__((ext_vector_type(16))) __bf16 v16bf;
typedef __attribute__((ext_vector_type(8)))  float  v8f;
typedef __attribute__((ext_vector_type(4)))  unsigned int v4u;
typedef __attribute__((ext_vector_type(8)))  int   v8i;
typedef __attribute__((ext_vector_type(4)))  int   v4i;

// Problem constants
#define NB   32
#define NC   64
#define NH   16
#define NW   16

// LDS layout
#define QS      1032                  // qpad per-channel stride (32x32 image + 8 pad -> bank spread)
#define QPAD_N  (16 * QS)             // bf16: 16 co channels, zero-padded 32x32 (interior at +8,+8)
#define KT_N    (256 * 64)            // bf16: kT[s][ck]   (s = p*16+q)
#define VTS     72                    // vT row stride (64 ck + 8 pad -> bank spread, 16B aligned)
#define VT_N    (1024 * VTS)          // bf16: vT[spatial(32x32 padded)][ck]  (interior at +7,+7)
#define ALPHA_N (2 * 1024)            // bf16: alphaP double buffer: [buf][ck/2][co][2]
#define XS_N    (64 * 16 * 16)        // f32: x[b] staged via TDM (raw copy)
#define WQ_OFF  0
#define WK_OFF  (16 * 36)
#define WV_OFF  (16 * 36 + 64 * 36)
#define WTS_N   (16 * 36 + 64 * 36 + 64 * 36)   // float weights
#define SMEM_BYTES ((size_t)(QPAD_N + KT_N + VT_N + ALPHA_N) * 2 + (size_t)(XS_N + WTS_N) * 4)

// 16-bit A-matrix (16x32) K index for VGPR i, pair pos, lane half (ISA 7.12.2)
__device__ __forceinline__ int kmapA(int i, int pos, int half) {
    return (i < 4 ? 2 * i : 16 + 2 * (i - 4)) + pos + 8 * half;
}
// 16-bit B-matrix (32x16) K index: same order as A (lanes 0-15 hold K=0..7,16..23; lanes 16-31 hold K=8..15,24..31)
__device__ __forceinline__ int kmapB(int i, int pos, int half) {
    return (i < 4 ? 2 * i : 16 + 2 * (i - 4)) + pos + 8 * half;
}

__global__ void __launch_bounds__(128) attn_dynconv_kernel(
    const float* __restrict__ xg,  const float* __restrict__ qwg,
    const float* __restrict__ kwg, const float* __restrict__ vwg,
    float* __restrict__ outg)
{
    extern __shared__ char smem_raw[];
    __bf16* qpad   = (__bf16*)smem_raw;
    __bf16* kT     = qpad + QPAD_N;
    __bf16* vT     = kT + KT_N;
    __bf16* alphaP = vT + VT_N;
    float*  xs     = (float*)(alphaP + ALPHA_N);  // 16-byte aligned region for TDM dest
    float*  wts    = xs + XS_N;

    const int tid  = threadIdx.x;
    const int wave = tid >> 5;
    const int lane = tid & 31;
    const int row  = lane & 15;   // A-row / B-col / D-col index
    const int half = lane >> 4;
    const int cotile = blockIdx.x;        // 0..3 : which 16 output channels
    const int b      = blockIdx.y;        // 0..31
    const int c0     = cotile * 16;

    // ---------------- Phase 0a: kick off TDM copy of x[b] (64 KB) global -> LDS ----------------
    if (wave == 0) {
        const unsigned long long ga =
            (unsigned long long)(const void*)(xg + (size_t)b * (NC * NH * NW));
        // LDS byte offset of xs: ptrtoint of an LDS-addrspace pointer
        const unsigned lds_off =
            (unsigned)(unsigned long long)(__attribute__((address_space(3))) const char*)(const void*)xs;
        // D# group 0 (ISA 8.3): count=1 | lds_addr | global_addr[56:0] | type=2
        v4u g0 = { 1u,
                   lds_off,
                   (unsigned)(ga & 0xFFFFFFFFu),
                   ((unsigned)((ga >> 32) & 0x01FFFFFFu)) | (2u << 30) };
        // D# group 1 (ISA 8.4): data_size=3 (8B elems); tensor 8192x1, tile 8192x1; stride0=8192
        v8i g1 = { (3 << 16),          // [17:16] data_size=3, mask/flags=0
                   (8192 << 16),       // [79:48] tensor_dim0 = 8192 (low 16 here)
                   (1 << 16),          // [79:64] dim0 hi = 0, [95:80] tensor_dim1 lo = 1
                   (8192 << 16),       // [127:112] dim1 hi = 0 | tile_dim0 = 8192
                   1,                  // tile_dim1 = 1, tile_dim2 = 0
                   8192,               // tensor_dim0_stride[31:0]
                   0, 0 };             // stride0 hi, tensor_dim1_stride
        v4i gz4 = { 0, 0, 0, 0 };
        v8i gz8 = { 0, 0, 0, 0, 0, 0, 0, 0 };
        __builtin_amdgcn_tensor_load_to_lds(g0, g1, gz4, gz4, gz8, 0);
    }

    // ---------------- Phase 0b: zero pads, stage weights (overlaps TDM) ----------------
    for (int i = tid; i < QPAD_N; i += 128) qpad[i] = (__bf16)0.0f;
    for (int i = tid; i < VT_N;   i += 128) vT[i]   = (__bf16)0.0f;
    for (int i = tid; i < 16 * 36; i += 128) wts[WQ_OFF + i] = qwg[c0 * 36 + i];
    for (int i = tid; i < 64 * 36; i += 128) wts[WK_OFF + i] = kwg[i];
    for (int i = tid; i < 64 * 36; i += 128) wts[WV_OFF + i] = vwg[i];
    if (wave == 0) __builtin_amdgcn_s_wait_tensorcnt(0);
    __syncthreads();

    // q (16 channels of this tile): 6x6 depthwise, pad left/top=2, right/bottom=3
    for (int e = tid; e < 16 * 256; e += 128) {
        const int ci = e >> 8, r = (e >> 4) & 15, cc = e & 15;
        float acc = 0.0f;
        #pragma unroll
        for (int i = 0; i < 6; ++i) {
            const int iy = r + i - 2;
            if (iy < 0 || iy >= 16) continue;
            #pragma unroll
            for (int j = 0; j < 6; ++j) {
                const int ix = cc + j - 2;
                if (ix < 0 || ix >= 16) continue;
                acc += xs[(c0 + ci) * 256 + iy * 16 + ix] * wts[WQ_OFF + ci * 36 + i * 6 + j];
            }
        }
        qpad[ci * QS + (r + 8) * 32 + (cc + 8)] = (__bf16)acc;
    }
    // k -> kT[s][ck] and v -> vT[spatial][ck] (all 64 channels); same taps, fused
    for (int e = tid; e < 64 * 256; e += 128) {
        const int ck = e >> 8, r = (e >> 4) & 15, cc = e & 15;
        float acck = 0.0f, accv = 0.0f;
        #pragma unroll
        for (int i = 0; i < 6; ++i) {
            const int iy = r + i - 2;
            if (iy < 0 || iy >= 16) continue;
            #pragma unroll
            for (int j = 0; j < 6; ++j) {
                const int ix = cc + j - 2;
                if (ix < 0 || ix >= 16) continue;
                const float xv = xs[ck * 256 + iy * 16 + ix];
                acck += xv * wts[WK_OFF + ck * 36 + i * 6 + j];
                accv += xv * wts[WV_OFF + ck * 36 + i * 6 + j];
            }
        }
        kT[(r * 16 + cc) * 64 + ck] = (__bf16)acck;
        vT[((r + 7) * 32 + (cc + 7)) * VTS + ck] = (__bf16)accv;
    }
    __syncthreads();

    // ---------------- Preload GEMM1 B-fragments (kT is (u,v)-invariant) ----------------
    v16bf b1[8];
    #pragma unroll
    for (int c = 0; c < 8; ++c) {
        #pragma unroll
        for (int j = 0; j < 16; ++j) {
            const int K = kmapB(j >> 1, j & 1, half);
            b1[c][j] = kT[(32 * c + K) * 64 + 16 * wave + row];
        }
    }

    v8f oaccT[4];   // transposed output tiles: D'[m=x, n=co] for y rows 4w..4w+3
    #pragma unroll
    for (int t = 0; t < 4; ++t)
        #pragma unroll
        for (int i = 0; i < 8; ++i) oaccT[t][i] = 0.0f;

    // ---------------- Main loop over 256 spatial offsets (u,v) ----------------
    #pragma unroll 1
    for (int uv = 0; uv < 256; ++uv) {
        const int u = uv >> 4, vv = uv & 15;
        __bf16* aph = alphaP + (uv & 1) * 1024;   // double-buffered alpha staging

        // GEMM1: alpha_uv[16 co x 64 ck] = Qshift(16x256) x kT(256x64); wave owns ck tile [16w,16w+16)
        v8f acc;
        #pragma unroll
        for (int i = 0; i < 8; ++i) acc[i] = 0.0f;
        #pragma unroll
        for (int c = 0; c < 8; ++c) {
            v16bf a1;
            #pragma unroll
            for (int j = 0; j < 16; ++j) {
                const int K = kmapA(j >> 1, j & 1, half);
                const int s = 32 * c + K;                   // s = p*16 + q
                a1[j] = qpad[row * QS + (u + (s >> 4)) * 32 + (vv + (s & 15))];
            }
            acc = __builtin_amdgcn_wmma_f32_16x16x32_bf16(
                      false, a1, false, b1[c], (short)0, acc, false, false);
        }
        // stage alpha transposed & pair-interleaved: aph[ck>>1][co][ck&1]
        // D layout: VGPR i -> co = 8*half+i, N col -> ck = 16*wave+row
        {
            const int ck = 16 * wave + row;
            const int base = (ck >> 1) * 32 + (ck & 1);
            #pragma unroll
            for (int i = 0; i < 8; ++i)
                aph[base + (8 * half + i) * 2] = (__bf16)acc[i];
        }
        __syncthreads();   // only barrier needed: next iter writes the other buffer

        // GEMM2 (transposed): outT[16 x(=m) x 16 co] += VsT(16x64) x alphaT(64x16)
        // B-fragments from aph: pairs along ck are adjacent -> b32 loads
        v16bf ba[2];
        #pragma unroll
        for (int c2 = 0; c2 < 2; ++c2) {
            #pragma unroll
            for (int j = 0; j < 16; ++j) {
                const int kk = 32 * c2 + kmapB(j >> 1, j & 1, half);  // K-dim = ck
                ba[c2][j] = aph[(kk >> 1) * 32 + row * 2 + (kk & 1)];
            }
        }
        #pragma unroll
        for (int t = 0; t < 4; ++t) {
            const int ty = 4 * wave + t;                    // output image row y
            const int spb = ((ty + u) * 32 + (row + vv)) * VTS;  // A row m = x = row
            #pragma unroll
            for (int c2 = 0; c2 < 2; ++c2) {
                v16bf av;   // A = VsT: rows contiguous in ck -> ds_load_b128 x2
                #pragma unroll
                for (int j = 0; j < 16; ++j) {
                    const int kk = 32 * c2 + kmapA(j >> 1, j & 1, half);
                    av[j] = vT[spb + kk];
                }
                oaccT[t] = __builtin_amdgcn_wmma_f32_16x16x32_bf16(
                               false, av, false, ba[c2], (short)0, oaccT[t], false, false);
            }
        }
    }

    // ---------------- Store output (B,C,H,W) f32 ----------------
    // D' layout: VGPR i -> x = 8*half+i, N col -> co_local = row; 8 consecutive floats per lane
    #pragma unroll
    for (int t = 0; t < 4; ++t) {
        const int y  = 4 * wave + t;
        const int co = c0 + row;
        #pragma unroll
        for (int i = 0; i < 8; ++i) {
            outg[((b * NC + co) * NH + y) * NW + (8 * half + i)] = oaccT[t][i];
        }
    }
}

extern "C" void kernel_launch(void* const* d_in, const int* in_sizes, int n_in,
                              void* d_out, int out_size, void* d_ws, size_t ws_size,
                              hipStream_t stream) {
    (void)in_sizes; (void)n_in; (void)d_ws; (void)ws_size; (void)out_size;
    const float* x  = (const float*)d_in[0];
    const float* qw = (const float*)d_in[1];
    const float* kw = (const float*)d_in[2];
    const float* vw = (const float*)d_in[3];
    float* out = (float*)d_out;

    // Allow ~297 KB dynamic LDS (WGP has 320 KB on CDNA5)
    (void)hipFuncSetAttribute(reinterpret_cast<const void*>(attn_dynconv_kernel),
                              hipFuncAttributeMaxDynamicSharedMemorySize, (int)SMEM_BYTES);

    dim3 grid(4, NB);   // 4 output-channel tiles x 32 batches = 128 workgroups
    dim3 block(128);    // 4 wave32 waves
    attn_dynconv_kernel<<<grid, block, SMEM_BYTES, stream>>>(x, qw, kw, vw, out);
}
